// VectorQuantizer_41042707481032
// MI455X (gfx1250) — matrix operand order repaired and checked by reading, not compiled, not run on hardware
//
#include <hip/hip_runtime.h>

// ---- CDNA5 WMMA vector types -------------------------------------------------
typedef __attribute__((ext_vector_type(16))) _Float16 v16h;
typedef __attribute__((ext_vector_type(8)))  _Float16 v8h;
typedef __attribute__((ext_vector_type(2)))  _Float16 v2h;
typedef __attribute__((ext_vector_type(8)))  float    v8f;
typedef __attribute__((ext_vector_type(4)))  float    v4f;

#define KCB     1024   // codebook entries
#define DDIM    64     // embedding dim
#define PITCH   72     // halves per codebook row in LDS (16B-aligned rows, bank-spread)
#define THREADS 256    // 8 wave32s
#define MROWS   64     // rows per wave (4 x 16-row WMMA sub-tiles sharing B)

__device__ __forceinline__ v16h load_a_frag(const float* xr, int kk, int hi) {
    // CDNA5 16-bit A 16x32 layout: lane(l<16): K=kk+0..7 (v0-3), kk+16..23 (v4-7);
    // lane(l>=16): +8 on both ranges.
    v4f p0 = *(const v4f*)(xr + kk + hi * 8);
    v4f p1 = *(const v4f*)(xr + kk + hi * 8 + 4);
    v4f p2 = *(const v4f*)(xr + kk + 16 + hi * 8);
    v4f p3 = *(const v4f*)(xr + kk + 16 + hi * 8 + 4);
    v16h a;
    a[0]=(_Float16)p0.x; a[1]=(_Float16)p0.y; a[2]=(_Float16)p0.z; a[3]=(_Float16)p0.w;
    a[4]=(_Float16)p1.x; a[5]=(_Float16)p1.y; a[6]=(_Float16)p1.z; a[7]=(_Float16)p1.w;
    a[8]=(_Float16)p2.x; a[9]=(_Float16)p2.y; a[10]=(_Float16)p2.z; a[11]=(_Float16)p2.w;
    a[12]=(_Float16)p3.x; a[13]=(_Float16)p3.y; a[14]=(_Float16)p3.z; a[15]=(_Float16)p3.w;
    return a;
}

__global__ __launch_bounds__(THREADS)
void VectorQuantizer_wmma_kernel(const float* __restrict__ x,
                                 const float* __restrict__ cb,
                                 float* __restrict__ outq,
                                 float* __restrict__ outidx,
                                 int nTiles64)
{
    // 144KB f16 codebook + 4KB norms -- fits CDNA5's 320KB-per-WGP LDS
    __shared__ _Float16 sCB[KCB * PITCH];
    __shared__ float    sEN[KCB];

    const int tid = threadIdx.x;

    // ---- Phase 1: stage codebook into LDS as f16 + compute ||e_k||^2 --------
    for (int r = tid; r < KCB; r += THREADS) {
        const v4f* row4 = (const v4f*)(cb + r * DDIM);
        float s = 0.f;
        #pragma unroll
        for (int j = 0; j < DDIM / 4; ++j) {
            v4f v = row4[j];
            s += v.x * v.x + v.y * v.y + v.z * v.z + v.w * v.w;
            v2h h01 = { (_Float16)v.x, (_Float16)v.y };
            v2h h23 = { (_Float16)v.z, (_Float16)v.w };
            *(v2h*)&sCB[r * PITCH + j * 4 + 0] = h01;
            *(v2h*)&sCB[r * PITCH + j * 4 + 2] = h23;
        }
        sEN[r] = s;
    }
    __syncthreads();

    const int wave = tid >> 5;
    const int lane = tid & 31;
    const int lh   = lane & 15;   // position within 16-lane half
    const int hi   = lane >> 4;   // 0: lanes 0-15, 1: lanes 16-31

    // One 64-row M-tile per wave: 512 blocks x 8 waves covers N/64 tiles exactly.
    const int tile = blockIdx.x * (THREADS / 32) + wave;
    if (tile >= nTiles64) return;
    const int row0 = tile * MROWS;

    // ---- A fragments for the four 16-row sub-tiles (they share B) ------------
    v16h a0[4], a1[4];
    #pragma unroll
    for (int st = 0; st < 4; ++st) {
        const float* xr = x + (size_t)(row0 + st * 16 + lh) * DDIM;
        a0[st] = load_a_frag(xr, 0,  hi);
        a1[st] = load_a_frag(xr, 32, hi);
    }

    float minv[4][8];
    int   mini[4][8];
    #pragma unroll
    for (int st = 0; st < 4; ++st)
        #pragma unroll
        for (int i = 0; i < 8; ++i) { minv[st][i] = 3.4e38f; mini[st][i] = 0; }

    // ---- Main loop: 64 codebook column-tiles, 8 WMMAs (4 indep chains) -------
    for (int kt = 0; kt < KCB / 16; ++kt) {
        // B 32x16 f16 layout: lanes 0-15 hold K(=d)=dk+0..7, dk+16..23 of column n=lh,
        // lanes 16-31 hold K=dk+8..15, dk+24..31 (matches A).
        const _Float16* bp = &sCB[(kt * 16 + lh) * PITCH + hi * 8];
        v8h bl0 = *(const v8h*)(bp);
        v8h bh0 = *(const v8h*)(bp + 16);
        v8h bl1 = *(const v8h*)(bp + 32);
        v8h bh1 = *(const v8h*)(bp + 48);
        v16h b0 = __builtin_shufflevector(bl0, bh0, 0,1,2,3,4,5,6,7,8,9,10,11,12,13,14,15);
        v16h b1 = __builtin_shufflevector(bl1, bh1, 0,1,2,3,4,5,6,7,8,9,10,11,12,13,14,15);

        v8f z = {0.f, 0.f, 0.f, 0.f, 0.f, 0.f, 0.f, 0.f};
        v8f c[4];
        // 4 independent accumulator chains; interleaved WMMAs hide RAW hazards.
        #pragma unroll
        for (int st = 0; st < 4; ++st)
            c[st] = __builtin_amdgcn_wmma_f32_16x16x32_f16(false, a0[st], false, b0,
                                                           (short)0, z, false, false);
        #pragma unroll
        for (int st = 0; st < 4; ++st)
            c[st] = __builtin_amdgcn_wmma_f32_16x16x32_f16(false, a1[st], false, b1,
                                                           (short)0, c[st], false, false);

        const float en   = sEN[kt * 16 + lh];
        const int   cand = kt * 16 + lh;
        #pragma unroll
        for (int st = 0; st < 4; ++st) {
            #pragma unroll
            for (int i = 0; i < 8; ++i) {
                float d = __builtin_fmaf(-2.f, c[st][i], en);   // ||e||^2 - 2 x.e
                bool tk = d < minv[st][i];
                minv[st][i] = tk ? d : minv[st][i];
                mini[st][i] = tk ? cand : mini[st][i];
            }
        }
    }

    // ---- Cross-lane argmin within 16-lane groups (branchless) ----------------
    #pragma unroll
    for (int m = 1; m < 16; m <<= 1) {
        #pragma unroll
        for (int st = 0; st < 4; ++st)
            #pragma unroll
            for (int i = 0; i < 8; ++i) {
                float ov = __shfl_xor(minv[st][i], m, 32);
                int   oi = __shfl_xor(mini[st][i], m, 32);
                bool take = (ov < minv[st][i]) |
                            ((ov == minv[st][i]) & (oi < mini[st][i]));
                minv[st][i] = take ? ov : minv[st][i];
                mini[st][i] = take ? oi : mini[st][i];
            }
    }

    // ---- Epilogue: gather exact fp32 codebook rows + write indices -----------
    #pragma unroll
    for (int st = 0; st < 4; ++st) {
        #pragma unroll
        for (int m = 0; m < 16; ++m) {
            int bi = __shfl(mini[st][m & 7], (m < 8) ? 0 : 16, 32);
            const int orow = row0 + st * 16 + m;
            float2 v = *(const float2*)(cb + (size_t)bi * DDIM + lane * 2);
            *(float2*)(outq + (size_t)orow * DDIM + lane * 2) = v;
            if (lane == 0) outidx[orow] = (float)bi;
        }
    }
}

extern "C" void kernel_launch(void* const* d_in, const int* in_sizes, int n_in,
                              void* d_out, int out_size, void* d_ws, size_t ws_size,
                              hipStream_t stream) {
    (void)d_ws; (void)ws_size; (void)out_size; (void)n_in;
    const float* x  = (const float*)d_in[0];      // (N, 64) f32
    const float* cb = (const float*)d_in[1];      // (1024, 64) f32
    const int N        = in_sizes[0] / DDIM;
    const int nTiles64 = N / MROWS;
    float* outq   = (float*)d_out;                // (N, 64) quantized
    float* outidx = outq + (size_t)N * DDIM;      // (N,) indices (as float)

    const int blocks = (nTiles64 + (THREADS / 32) - 1) / (THREADS / 32);
    VectorQuantizer_wmma_kernel<<<blocks, THREADS, 0, stream>>>(x, cb, outq, outidx, nTiles64);
}
